// CSPDarknet_33672543601012
// MI455X (gfx1250) — compile-verified, matrix operand order checked by reading, not run on hardware
//
#include <hip/hip_runtime.h>
#include <hip/hip_bf16.h>
#include <cstdint>

typedef __attribute__((ext_vector_type(16))) _Float16 v16h;
typedef __attribute__((ext_vector_type(8)))  float    v8f;

#define D_MODEL 192
#define D_STATE 16
#define D_INNER 384
#define DT_RANK 12
#define KDIR    4
#define BATCH   2
#define HH      80
#define WW      80
#define LTOT    (HH*WW)          // 6400
#define MROWS   (BATCH*LTOT)     // 12800
#define CPROJ   48               // 44 padded to 48 for 16x16 tiles

// ---------------------------------------------------------------------------
// direction permutation: scan-step index l -> row-major spatial index p
//   k=0: l      k=1: (w,h)-major      k=2: reversed      k=3: reversed (w,h)
// ---------------------------------------------------------------------------
__device__ __forceinline__ int perm_idx(int k, int l) {
    int lm = (k & 2) ? (LTOT - 1 - l) : l;
    if (k & 1) { int hh = lm % HH; int ww = lm / HH; return hh * WW + ww; }
    return lm;
}

__device__ __forceinline__ float sigmoidf_(float x) { return 1.0f / (1.0f + __expf(-x)); }

__device__ __forceinline__ float wave_sum(float s) {
#pragma unroll
    for (int o = 16; o > 0; o >>= 1) s += __shfl_xor(s, o, 32);
    return s;
}

// ---------------------------------------------------------------------------
// f32 -> f16 weight casts
// ---------------------------------------------------------------------------
__global__ void k_cvt_f16(const float* __restrict__ s, _Float16* __restrict__ d, int n) {
    int i = blockIdx.x * blockDim.x + threadIdx.x;
    if (i < n) d[i] = (_Float16)s[i];
}

// x_proj_weight (K,44,384) -> padded f16 (K,48,384)
__global__ void k_pad_xproj(const float* __restrict__ s, _Float16* __restrict__ d) {
    int i = blockIdx.x * blockDim.x + threadIdx.x;
    if (i >= KDIR * CPROJ * D_INNER) return;
    int din = i % D_INNER;
    int c   = (i / D_INNER) % CPROJ;
    int k   = i / (CPROJ * D_INNER);
    d[i] = (c < (DT_RANK + 2 * D_STATE))
         ? (_Float16)s[((size_t)k * (DT_RANK + 2 * D_STATE) + c) * D_INNER + din]
         : (_Float16)0.0f;
}

__global__ void k_zero_f32(float* __restrict__ d, int n) {
    int i = blockIdx.x * blockDim.x + threadIdx.x;
    if (i < n) d[i] = 0.0f;
}

// ---------------------------------------------------------------------------
// input layernorm, one wave per (b,l) row (6 elems/lane), emit f16
// ---------------------------------------------------------------------------
__global__ void k_ln_in(const float* __restrict__ x, const float* __restrict__ w,
                        const float* __restrict__ b, _Float16* __restrict__ out) {
    const int lane = threadIdx.x & 31;
    const int r = blockIdx.x * (blockDim.x >> 5) + (threadIdx.x >> 5);
    if (r >= MROWS) return;
    const float* px = x + (size_t)r * D_MODEL;
    float vals[6];
    float s = 0.f;
#pragma unroll
    for (int i = 0; i < 6; ++i) { vals[i] = px[lane + 32 * i]; s += vals[i]; }
    const float m = wave_sum(s) * (1.0f / D_MODEL);
    float v = 0.f;
#pragma unroll
    for (int i = 0; i < 6; ++i) { float d0 = vals[i] - m; v += d0 * d0; }
    const float rs = rsqrtf(wave_sum(v) * (1.0f / D_MODEL) + 1e-5f);
    _Float16* po = out + (size_t)r * D_MODEL;
#pragma unroll
    for (int i = 0; i < 6; ++i) {
        int c = lane + 32 * i;
        po[c] = (_Float16)((vals[i] - m) * rs * w[c] + b[c]);
    }
}

// ---------------------------------------------------------------------------
// in_proj GEMM: out[m,n] = sum_k Xn[m,k] * W[n,k]   (M=12800, N=768, K=192)
// one wave computes a 16x64 tile (A fragment reused across 4 B fragments).
// N-blocks of 64 align with the xh/z split at n=384 -> store branch is uniform.
// ---------------------------------------------------------------------------
__global__ void k_gemm_inproj(const _Float16* __restrict__ Xn, const _Float16* __restrict__ Wh,
                              float* __restrict__ xh, float* __restrict__ z) {
    const int lane = threadIdx.x & 31;
    const int n0 = blockIdx.x * 64;
    const int m0 = blockIdx.y * 16;
    const int mm = lane & 15;
    const int hi8 = (lane >> 4) ? 8 : 0;
    v8f acc[4] = {};
    for (int k0 = 0; k0 < D_MODEL; k0 += 32) {
        v16h a;
        const _Float16* pa = Xn + (size_t)(m0 + mm) * D_MODEL + k0 + hi8;
#pragma unroll
        for (int j = 0; j < 8; ++j) { a[j] = pa[j]; a[8 + j] = pa[16 + j]; }
#pragma unroll
        for (int t = 0; t < 4; ++t) {
            v16h bf;
            const _Float16* pb = Wh + (size_t)(n0 + 16 * t + mm) * D_MODEL + k0 + hi8;
#pragma unroll
            for (int j = 0; j < 8; ++j) { bf[j] = pb[j]; bf[8 + j] = pb[16 + j]; }
            acc[t] = __builtin_amdgcn_wmma_f32_16x16x32_f16(false, a, false, bf,
                                                            (short)0, acc[t], false, false);
        }
    }
    const int bb = m0 / LTOT;                 // 16-row tile never crosses a batch
    const int lbase = m0 - bb * LTOT;
    if (n0 < D_INNER) {                       // wave-uniform: whole 64-block in xh
#pragma unroll
        for (int t = 0; t < 4; ++t) {
            int nn = n0 + 16 * t + mm;
#pragma unroll
            for (int r = 0; r < 8; ++r)
                xh[((size_t)bb * D_INNER + nn) * LTOT + lbase + r + hi8] = acc[t][r];
        }
    } else {                                  // whole 64-block in z
#pragma unroll
        for (int t = 0; t < 4; ++t) {
            int nn = n0 + 16 * t + mm - D_INNER;
#pragma unroll
            for (int r = 0; r < 8; ++r)
                z[(size_t)(m0 + r + hi8) * D_INNER + nn] = acc[t][r];
        }
    }
}

// ---------------------------------------------------------------------------
// depthwise 3x3 conv (SAME) + bias + SiLU on (B, D_INNER, H, W).
// outputs stored TRANSPOSED (B, L, D_INNER): makes the x_dbl WMMA B-fragment
// K-dimension contiguous and the scan u-loads wave-coalesced.
// ---------------------------------------------------------------------------
__global__ void k_dwconv(const float* __restrict__ xh, const float* __restrict__ cw,
                         const float* __restrict__ cb, float* __restrict__ xcT,
                         _Float16* __restrict__ xchT) {
    int i = blockIdx.x * blockDim.x + threadIdx.x;
    if (i >= BATCH * D_INNER * LTOT) return;
    int l = i % LTOT;
    int d = (i / LTOT) % D_INNER;
    int b = i / (LTOT * D_INNER);
    int h = l / WW, w = l % WW;
    const float* src = xh + ((size_t)b * D_INNER + d) * LTOT;
    const float* wk  = cw + (size_t)d * 9;
    float s = cb[d];
#pragma unroll
    for (int kh = 0; kh < 3; ++kh) {
        int ih = h + kh - 1;
        if (ih < 0 || ih >= HH) continue;
#pragma unroll
        for (int kw = 0; kw < 3; ++kw) {
            int iw = w + kw - 1;
            if (iw < 0 || iw >= WW) continue;
            s = fmaf(src[ih * WW + iw], wk[kh * 3 + kw], s);
        }
    }
    s = s * sigmoidf_(s);            // SiLU
    size_t o = ((size_t)b * LTOT + l) * D_INNER + d;
    xcT[o]  = s;
    xchT[o] = (_Float16)s;
}

// ---------------------------------------------------------------------------
// x_dbl GEMM per (b,k): out (48 x L) = Wp_k (48x384) @ xs_k (384xL).
// One wave computes a full 48x16 column block: B fragment (direction-permuted
// view of transposed conv output, contiguous in K) loaded once, reused for
// all 3 M-tiles of the weight.
// ---------------------------------------------------------------------------
__global__ void k_gemm_xdbl(const _Float16* __restrict__ Wp, const _Float16* __restrict__ XchT,
                            float* __restrict__ xdbl) {
    const int lane = threadIdx.x & 31;
    const int l0 = blockIdx.x * 16;
    const int bk = blockIdx.y;                 // b*KDIR + k
    const int b = bk / KDIR, k = bk % KDIR;
    const int mm = lane & 15;
    const int hi8 = (lane >> 4) ? 8 : 0;
    const _Float16* Wk = Wp + (size_t)k * CPROJ * D_INNER;
    const int pcol = perm_idx(k, l0 + mm);     // this lane's column in scan order
    const _Float16* bbase = XchT + ((size_t)b * LTOT + pcol) * D_INNER + hi8;
    v8f acc[3] = {};
    for (int k0 = 0; k0 < D_INNER; k0 += 32) {
        v16h bf;
        const _Float16* pb = bbase + k0;
#pragma unroll
        for (int j = 0; j < 8; ++j) { bf[j] = pb[j]; bf[8 + j] = pb[16 + j]; }
#pragma unroll
        for (int t = 0; t < 3; ++t) {
            v16h a;
            const _Float16* pa = Wk + (size_t)(16 * t + mm) * D_INNER + k0 + hi8;
#pragma unroll
            for (int j = 0; j < 8; ++j) { a[j] = pa[j]; a[8 + j] = pa[16 + j]; }
            acc[t] = __builtin_amdgcn_wmma_f32_16x16x32_f16(false, a, false, bf,
                                                            (short)0, acc[t], false, false);
        }
    }
    float* outb = xdbl + (size_t)bk * CPROJ * LTOT;
#pragma unroll
    for (int t = 0; t < 3; ++t)
#pragma unroll
        for (int r = 0; r < 8; ++r)
            outb[(size_t)(16 * t + r + hi8) * LTOT + l0 + mm] = acc[t][r];
}

// ---------------------------------------------------------------------------
// selective scan: one lane per (b,k,d). State h[16] in VGPRs; per-step dts/B/C
// are wave-uniform (scalar loads through the constant cache). u-loads and the
// atomic direction-merge are coalesced via the (B,L,D) layout.
// ---------------------------------------------------------------------------
__global__ void k_scan(const float* __restrict__ xdbl, const float* __restrict__ xcT,
                       const float* __restrict__ dtw, const float* __restrict__ dtb,
                       const float* __restrict__ A_logs, const float* __restrict__ Ds,
                       float* __restrict__ ym) {
    const int d  = blockIdx.x * blockDim.x + threadIdx.x;   // 0..383
    const int bk = blockIdx.y;                              // b*KDIR + k
    if (d >= D_INNER) return;
    const int b = bk / KDIR, k = bk % KDIR;
    const int kd = k * D_INNER + d;

    float Arow[D_STATE];
#pragma unroll
    for (int n = 0; n < D_STATE; ++n)
        Arow[n] = -__expf(A_logs[(size_t)kd * D_STATE + n]);
    float dwr[DT_RANK];
#pragma unroll
    for (int r = 0; r < DT_RANK; ++r)
        dwr[r] = dtw[(size_t)kd * DT_RANK + r];
    const float bias = dtb[kd];
    const float Dv   = Ds[kd];

    float h[D_STATE];
#pragma unroll
    for (int n = 0; n < D_STATE; ++n) h[n] = 0.0f;

    const float* xd  = xdbl + (size_t)bk * CPROJ * LTOT;    // wave-uniform base
    const float* xcb = xcT + (size_t)b * LTOT * D_INNER + d;
    float* ymb = ym + (size_t)b * LTOT * D_INNER + d;

    for (int l = 0; l < LTOT; ++l) {
        const int p = perm_idx(k, l);
        const float u = xcb[(size_t)p * D_INNER];           // coalesced across lanes
        // dt projection (wave-uniform dts, per-lane weights)
        float s = bias;
#pragma unroll
        for (int r = 0; r < DT_RANK; ++r)
            s = fmaf(dwr[r], xd[(size_t)r * LTOT + l], s);
        const float delta = (s > 20.0f) ? s : __logf(1.0f + __expf(s));  // softplus
        const float du = delta * u;
        float y = 0.0f;
#pragma unroll
        for (int n = 0; n < D_STATE; ++n) {
            const float Bn = xd[(size_t)(DT_RANK + n)           * LTOT + l];
            const float Cn = xd[(size_t)(DT_RANK + D_STATE + n) * LTOT + l];
            h[n] = fmaf(h[n], __expf(delta * Arow[n]), du * Bn);
            y = fmaf(h[n], Cn, y);
        }
        atomicAdd(&ymb[(size_t)p * D_INNER], y + Dv * u);   // merged at row-major p
    }
}

// ---------------------------------------------------------------------------
// out layernorm + SiLU(z) gate, one wave per (b,l) row (12 elems/lane)
// ---------------------------------------------------------------------------
__global__ void k_ln_gate(const float* __restrict__ ym, const float* __restrict__ w,
                          const float* __restrict__ b, const float* __restrict__ z,
                          _Float16* __restrict__ yg) {
    const int lane = threadIdx.x & 31;
    const int r = blockIdx.x * (blockDim.x >> 5) + (threadIdx.x >> 5);
    if (r >= MROWS) return;
    const float* py = ym + (size_t)r * D_INNER;
    const float* pz = z  + (size_t)r * D_INNER;
    float vals[12];
    float s = 0.f;
#pragma unroll
    for (int i = 0; i < 12; ++i) { vals[i] = py[lane + 32 * i]; s += vals[i]; }
    const float m = wave_sum(s) * (1.0f / D_INNER);
    float v = 0.f;
#pragma unroll
    for (int i = 0; i < 12; ++i) { float d0 = vals[i] - m; v += d0 * d0; }
    const float rs = rsqrtf(wave_sum(v) * (1.0f / D_INNER) + 1e-5f);
    _Float16* po = yg + (size_t)r * D_INNER;
#pragma unroll
    for (int i = 0; i < 12; ++i) {
        int c = lane + 32 * i;
        float ln = (vals[i] - m) * rs * w[c] + b[c];
        float zz = pz[c];
        po[c] = (_Float16)(ln * zz * sigmoidf_(zz));
    }
}

// ---------------------------------------------------------------------------
// out_proj GEMM: out[m,n] = sum_c yg[m,c] * Wout[n,c]  (M=12800, N=192, K=384)
// one wave computes a 16x64 tile (A reused across 4 B fragments).
// ---------------------------------------------------------------------------
__global__ void k_gemm_outproj(const _Float16* __restrict__ Yg, const _Float16* __restrict__ Wh,
                               float* __restrict__ out) {
    const int lane = threadIdx.x & 31;
    const int n0 = blockIdx.x * 64;
    const int m0 = blockIdx.y * 16;
    const int mm = lane & 15;
    const int hi8 = (lane >> 4) ? 8 : 0;
    v8f acc[4] = {};
    for (int k0 = 0; k0 < D_INNER; k0 += 32) {
        v16h a;
        const _Float16* pa = Yg + (size_t)(m0 + mm) * D_INNER + k0 + hi8;
#pragma unroll
        for (int j = 0; j < 8; ++j) { a[j] = pa[j]; a[8 + j] = pa[16 + j]; }
#pragma unroll
        for (int t = 0; t < 4; ++t) {
            v16h bf;
            const _Float16* pb = Wh + (size_t)(n0 + 16 * t + mm) * D_INNER + k0 + hi8;
#pragma unroll
            for (int j = 0; j < 8; ++j) { bf[j] = pb[j]; bf[8 + j] = pb[16 + j]; }
            acc[t] = __builtin_amdgcn_wmma_f32_16x16x32_f16(false, a, false, bf,
                                                            (short)0, acc[t], false, false);
        }
    }
#pragma unroll
    for (int t = 0; t < 4; ++t)
#pragma unroll
        for (int r = 0; r < 8; ++r)
            out[(size_t)(m0 + r + hi8) * D_MODEL + n0 + 16 * t + mm] = acc[t][r];
}

// ---------------------------------------------------------------------------
extern "C" void kernel_launch(void* const* d_in, const int* in_sizes, int n_in,
                              void* d_out, int out_size, void* d_ws, size_t ws_size,
                              hipStream_t stream) {
    (void)in_sizes; (void)n_in; (void)out_size; (void)ws_size;
    const float* x        = (const float*)d_in[0];
    const float* ln_w     = (const float*)d_in[1];
    const float* ln_b     = (const float*)d_in[2];
    const float* in_pw    = (const float*)d_in[3];
    const float* conv_w   = (const float*)d_in[4];
    const float* conv_b   = (const float*)d_in[5];
    const float* x_proj_w = (const float*)d_in[6];
    const float* dt_w     = (const float*)d_in[7];
    const float* dt_b     = (const float*)d_in[8];
    const float* A_logs   = (const float*)d_in[9];
    const float* Ds       = (const float*)d_in[10];
    const float* on_w     = (const float*)d_in[11];
    const float* on_b     = (const float*)d_in[12];
    const float* out_pw   = (const float*)d_in[13];
    float* out = (float*)d_out;

    char* p = (char*)d_ws;
    auto alloc = [&](size_t bytes) -> void* {
        void* r = (void*)p;
        p += (bytes + 255) & ~(size_t)255;
        return r;
    };
    _Float16* xn_h  = (_Float16*)alloc((size_t)MROWS * D_MODEL * 2);
    _Float16* win_h = (_Float16*)alloc((size_t)2 * D_INNER * D_MODEL * 2);
    _Float16* wxp_h = (_Float16*)alloc((size_t)KDIR * CPROJ * D_INNER * 2);
    _Float16* wout_h= (_Float16*)alloc((size_t)D_MODEL * D_INNER * 2);
    float*    xh    = (float*)   alloc((size_t)BATCH * D_INNER * LTOT * 4);
    float*    z     = (float*)   alloc((size_t)MROWS * D_INNER * 4);
    float*    xcT   = (float*)   alloc((size_t)BATCH * LTOT * D_INNER * 4);
    _Float16* xchT  = (_Float16*)alloc((size_t)BATCH * LTOT * D_INNER * 2);
    float*    xdbl  = (float*)   alloc((size_t)BATCH * KDIR * CPROJ * LTOT * 4);
    float*    ym    = (float*)   alloc((size_t)MROWS * D_INNER * 4);
    _Float16* yg_h  = (_Float16*)alloc((size_t)MROWS * D_INNER * 2);

    // weight casts
    {
        int n1 = 2 * D_INNER * D_MODEL;
        k_cvt_f16<<<(n1 + 255) / 256, 256, 0, stream>>>(in_pw, win_h, n1);
        int n2 = KDIR * CPROJ * D_INNER;
        k_pad_xproj<<<(n2 + 255) / 256, 256, 0, stream>>>(x_proj_w, wxp_h);
        int n3 = D_MODEL * D_INNER;
        k_cvt_f16<<<(n3 + 255) / 256, 256, 0, stream>>>(out_pw, wout_h, n3);
    }
    // input LN -> f16 (wave per row, 8 rows per 256-thread block)
    k_ln_in<<<(MROWS + 7) / 8, 256, 0, stream>>>(x, ln_w, ln_b, xn_h);
    // in_proj WMMA GEMM (16x64 tiles)
    k_gemm_inproj<<<dim3((2 * D_INNER) / 64, MROWS / 16), 32, 0, stream>>>(xn_h, win_h, xh, z);
    // depthwise conv + SiLU -> transposed (B,L,D) f32 + f16
    {
        int n = BATCH * D_INNER * LTOT;
        k_dwconv<<<(n + 255) / 256, 256, 0, stream>>>(xh, conv_w, conv_b, xcT, xchT);
    }
    // x_dbl WMMA GEMM (48x16 per wave; direction permutation folded into B loads)
    k_gemm_xdbl<<<dim3(LTOT / 16, BATCH * KDIR), 32, 0, stream>>>(wxp_h, xchT, xdbl);
    // zero merged output, then scan + merge
    {
        int n = MROWS * D_INNER;
        k_zero_f32<<<(n + 255) / 256, 256, 0, stream>>>(ym, n);
    }
    k_scan<<<dim3(D_INNER / 32, BATCH * KDIR), 32, 0, stream>>>(
        xdbl, xcT, dt_w, dt_b, A_logs, Ds, ym);
    // out LN + SiLU gate -> f16 (wave per row)
    k_ln_gate<<<(MROWS + 7) / 8, 256, 0, stream>>>(ym, on_w, on_b, z, yg_h);
    // out_proj WMMA GEMM -> d_out (16x64 tiles)
    k_gemm_outproj<<<dim3(D_MODEL / 64, MROWS / 16), 32, 0, stream>>>(yg_h, wout_h, out);
}
